// GQA_65927747993825
// MI455X (gfx1250) — matrix operand order repaired and checked by reading, not compiled, not run on hardware
//
#include <hip/hip_runtime.h>
#include <hip/hip_bf16.h>

typedef __attribute__((ext_vector_type(16))) _Float16 v16h;
typedef __attribute__((ext_vector_type(8)))  float    v8f;
typedef __attribute__((ext_vector_type(4)))  float    v4f;

namespace {
constexpr int BB  = 2;
constexpr int SS  = 2048;
constexpr int EE  = 1024;
constexpr int HH  = 16;
constexpr int GG  = 4;
constexpr int HSZ = 64;
constexpr int MT  = BB * SS;     // 4096 flattened rows
constexpr int KVE = GG * HSZ;    // 256
constexpr int KT  = 64;          // keys per attention iteration
// softmax scale (64^-0.5) pre-multiplied by log2(e): run softmax in exp2 domain
constexpr float QSCALE_LOG2E = 0.125f * 1.44269504088896340736f;
}

// 16 f16 values per lane; two 16B chunks
union Frag {
  v16h v;
  v4f  f4[2];
};

// A-matrix fragment (16xK tile of a row-major [rows x ld] f16 matrix).
// ISA layout: lanes 0-15 hold M=lane, K={0..7,16..23}; lanes 16-31 hold
// M=lane-16, K={8..15,24..31}.
__device__ __forceinline__ Frag load_a(const _Float16* base, int ld, int m0,
                                       int k0, int lane) {
  int row = m0 + (lane & 15);
  int kb  = (lane < 16) ? 0 : 8;
  const _Float16* p = base + (size_t)row * ld + k0;
  Frag f;
  f.f4[0] = *(const v4f*)(p + kb);
  f.f4[1] = *(const v4f*)(p + 16 + kb);
  return f;
}

// B-matrix fragment: B[k,n] with B^T stored row-major (source matrix [N x ld],
// K contiguous). Lanes 0-15 hold N=lane, K={0..7,16..23}; lanes 16-31 hold
// N=lane-16, K={8..15,24..31}.
__device__ __forceinline__ Frag load_b(const _Float16* base, int ld, int n0,
                                       int k0, int lane) {
  int col = n0 + (lane & 15);
  int kg  = (lane < 16) ? 0 : 8;
  const _Float16* p = base + (size_t)col * ld + k0 + kg;
  Frag f;
  f.f4[0] = *(const v4f*)(p);
  f.f4[1] = *(const v4f*)(p + 16);
  return f;
}

__device__ __forceinline__ v8f wmma16(const Frag& a, const Frag& b, v8f c) {
  return __builtin_amdgcn_wmma_f32_16x16x32_f16(false, a.v, false, b.v,
                                                (short)0, c, false, false);
}

__device__ __forceinline__ float half16_max(float v) {
  v = fmaxf(v, __shfl_xor(v, 8, 32));
  v = fmaxf(v, __shfl_xor(v, 4, 32));
  v = fmaxf(v, __shfl_xor(v, 2, 32));
  v = fmaxf(v, __shfl_xor(v, 1, 32));
  return v;
}

// ---------------------------------------------------------------- convert
__global__ void f32_to_f16_kernel(const float* __restrict__ in,
                                  _Float16* __restrict__ out, int n) {
  int i = blockIdx.x * blockDim.x + threadIdx.x;
  int stride = gridDim.x * blockDim.x;
  for (; i < n; i += stride) out[i] = (_Float16)in[i];
}

// ---------------------------------------------------------------- GEMM
// C[M,N] = A[M,K] @ W[N,K]^T ; each wave computes a 32x64 strip
// (2 M-tiles x 4 N-tiles, B fragments shared across the M-tiles).
// OUTMODE 0: f16 row-major; 1: f16 transposed [N,M]; 2: f32 row-major + bias.
template <int OUTMODE>
__global__ __launch_bounds__(256) void gemm_nt_kernel(
    const _Float16* __restrict__ A, const _Float16* __restrict__ W,
    _Float16* __restrict__ Ch, float* __restrict__ Cf,
    const float* __restrict__ bias, int M, int N, int K) {
  const int lane = threadIdx.x & 31;
  const int wave = blockIdx.x * (blockDim.x >> 5) + (threadIdx.x >> 5);
  const int tilesN = N >> 6;
  const int mt = wave / tilesN;
  const int nt = wave % tilesN;
  if (mt * 32 >= M) return;
  const int m0 = mt * 32, n0 = nt * 64;

  v8f acc[2][4];
#pragma unroll
  for (int mi = 0; mi < 2; ++mi)
#pragma unroll
    for (int t = 0; t < 4; ++t) acc[mi][t] = (v8f)0.0f;

  for (int k0 = 0; k0 < K; k0 += 32) {
    Frag fa0 = load_a(A, K, m0, k0, lane);
    Frag fa1 = load_a(A, K, m0 + 16, k0, lane);
#pragma unroll
    for (int t = 0; t < 4; ++t) {
      Frag fb = load_b(W, K, n0 + 16 * t, k0, lane);
      acc[0][t] = wmma16(fa0, fb, acc[0][t]);
      acc[1][t] = wmma16(fa1, fb, acc[1][t]);
    }
  }

#pragma unroll
  for (int mi = 0; mi < 2; ++mi) {
#pragma unroll
    for (int j = 0; j < 8; ++j) {
      int row = m0 + 16 * mi + j + ((lane >> 4) << 3);
      int col = n0 + (lane & 15);
      if (OUTMODE == 0) {
        _Float16* p = Ch + (size_t)row * N + col;
#pragma unroll
        for (int t = 0; t < 4; ++t) p[16 * t] = (_Float16)acc[mi][t][j];
      } else if (OUTMODE == 1) {
#pragma unroll
        for (int t = 0; t < 4; ++t)
          Ch[(size_t)(col + 16 * t) * M + row] = (_Float16)acc[mi][t][j];
      } else {
        float* p = Cf + (size_t)row * N + col;
#pragma unroll
        for (int t = 0; t < 4; ++t) p[16 * t] = acc[mi][t][j] + bias[col + 16 * t];
      }
    }
  }
}

// ---------------------------------------------------------------- attention
// One wave per (b, h, 16-row q tile); streams 64 keys / iteration with online
// softmax run entirely in the exp2 domain (scale * log2e folded into the Q
// fragments, raw v_exp_f32 for probabilities and corrections). Row-sum is
// fused into a 5th WMMA accumulator against a ones B-fragment.
// Q: [MT,EE] f16 (head h at cols h*64). K: [MT,KVE] f16.
// Vt: [KVE, MT] f16 (d-major so P@V B-fragments read keys contiguously).
__global__ __launch_bounds__(128) void gqa_attn_kernel(
    const _Float16* __restrict__ Qh, const _Float16* __restrict__ Kh,
    const _Float16* __restrict__ Vt, _Float16* __restrict__ Ctx) {
  __shared__ _Float16 Pl[4][16 * KT];  // 2KB per wave P staging (C->A relayout)

  const int lane = threadIdx.x & 31;
  const int ws   = threadIdx.x >> 5;
  const int wave = blockIdx.x * 4 + ws;

  const int qt = wave & 127;          // SS/16 = 128 q tiles
  const int bh = wave >> 7;
  const int h  = bh & (HH - 1);
  const int b  = bh >> 4;
  const int g  = h >> 2;              // group = h / (H/G)

  const _Float16* qb  = Qh + (size_t)b * SS * EE + (size_t)h * HSZ;
  const _Float16* kb  = Kh + (size_t)b * SS * KVE + (size_t)g * HSZ;
  const _Float16* vtb = Vt + (size_t)(g * HSZ) * MT + (size_t)b * SS;

  const int q0 = qt * 16;
  Frag aq0 = load_a(qb, EE, q0, 0, lane);
  Frag aq1 = load_a(qb, EE, q0, 32, lane);
  aq0.v = aq0.v * (_Float16)QSCALE_LOG2E;  // scores come out in log2 units
  aq1.v = aq1.v * (_Float16)QSCALE_LOG2E;

  Frag ones;
#pragma unroll
  for (int i = 0; i < 16; ++i) ones.v[i] = (_Float16)1.0f;

  float mrow[8];
  v8f o[4];
  v8f lacc = (v8f)0.0f;
#pragma unroll
  for (int j = 0; j < 8; ++j) mrow[j] = -1e30f;
#pragma unroll
  for (int t = 0; t < 4; ++t) o[t] = (v8f)0.0f;

  for (int kk = 0; kk < SS; kk += KT) {
    // scores: 16 q rows x 64 keys (4 n-tiles x 2 d-steps = 8 WMMAs)
    v8f c[4];
#pragma unroll
    for (int t = 0; t < 4; ++t) {
      Frag bk0 = load_b(kb, KVE, kk + 16 * t, 0, lane);
      Frag bk1 = load_b(kb, KVE, kk + 16 * t, 32, lane);
      c[t] = wmma16(aq0, bk0, (v8f)0.0f);
      c[t] = wmma16(aq1, bk1, c[t]);
    }

    // online softmax per row (rows live in 16-lane halves of each acc VGPR)
#pragma unroll
    for (int j = 0; j < 8; ++j) {
      float mx = fmaxf(fmaxf(c[0][j], c[1][j]), fmaxf(c[2][j], c[3][j]));
      mx = half16_max(mx);
      float mn = fmaxf(mrow[j], mx);
      float corr = __builtin_amdgcn_exp2f(mrow[j] - mn);
      mrow[j] = mn;
      float p0 = __builtin_amdgcn_exp2f(c[0][j] - mn);
      float p1 = __builtin_amdgcn_exp2f(c[1][j] - mn);
      float p2 = __builtin_amdgcn_exp2f(c[2][j] - mn);
      float p3 = __builtin_amdgcn_exp2f(c[3][j] - mn);
      o[0][j] *= corr; o[1][j] *= corr; o[2][j] *= corr; o[3][j] *= corr;
      lacc[j] *= corr;
      int prow = j + ((lane >> 4) << 3);
      _Float16* pr = &Pl[ws][prow * KT + (lane & 15)];
      pr[0]  = (_Float16)p0;
      pr[16] = (_Float16)p1;
      pr[32] = (_Float16)p2;
      pr[48] = (_Float16)p3;
    }
    asm volatile("s_wait_dscnt 0x0" ::: "memory");

    // P (16x64) as two A-fragments
    Frag ap0 = load_a(&Pl[ws][0], KT, 0, 0, lane);
    Frag ap1 = load_a(&Pl[ws][0], KT, 0, 32, lane);

    // row-sum of P via ones B-fragment (2 WMMAs, replaces shuffle reduction)
    lacc = wmma16(ap0, ones, lacc);
    lacc = wmma16(ap1, ones, lacc);

    // P @ V tile (64 keys x 64 d): 4 n-tiles x 2 k-steps = 8 WMMAs
#pragma unroll
    for (int t = 0; t < 4; ++t) {
      Frag bv0 = load_b(vtb, MT, 16 * t, kk, lane);
      Frag bv1 = load_b(vtb, MT, 16 * t, kk + 32, lane);
      o[t] = wmma16(ap0, bv0, o[t]);
      o[t] = wmma16(ap1, bv1, o[t]);
    }
  }

  // normalize and write context rows (f16, [MT, EE] head-major)
#pragma unroll
  for (int j = 0; j < 8; ++j) {
    float inv = 1.0f / lacc[j];
    int row = q0 + j + ((lane >> 4) << 3);
    _Float16* p = Ctx + ((size_t)b * SS + row) * EE + h * HSZ + (lane & 15);
    p[0]  = (_Float16)(o[0][j] * inv);
    p[16] = (_Float16)(o[1][j] * inv);
    p[32] = (_Float16)(o[2][j] * inv);
    p[48] = (_Float16)(o[3][j] * inv);
  }
}

// ---------------------------------------------------------------- launch
extern "C" void kernel_launch(void* const* d_in, const int* in_sizes, int n_in,
                              void* d_out, int out_size, void* d_ws,
                              size_t ws_size, hipStream_t stream) {
  (void)in_sizes; (void)n_in; (void)out_size; (void)ws_size;
  const float* x  = (const float*)d_in[0];
  const float* Wq = (const float*)d_in[1];
  const float* Wk = (const float*)d_in[2];
  const float* Wv = (const float*)d_in[3];
  const float* Wo = (const float*)d_in[4];
  const float* bo = (const float*)d_in[5];

  // workspace layout (f16 elements); total ~33 MB
  _Float16* xh  = (_Float16*)d_ws;
  _Float16* wqh = xh  + (size_t)MT * EE;
  _Float16* wkh = wqh + (size_t)EE * EE;
  _Float16* wvh = wkh + (size_t)KVE * EE;
  _Float16* woh = wvh + (size_t)KVE * EE;
  _Float16* qh  = woh + (size_t)EE * EE;
  _Float16* kh  = qh  + (size_t)MT * EE;
  _Float16* vt  = kh  + (size_t)MT * KVE;
  _Float16* ctx = vt  + (size_t)MT * KVE;

  f32_to_f16_kernel<<<2048, 256, 0, stream>>>(x,  xh,  MT * EE);
  f32_to_f16_kernel<<<512,  256, 0, stream>>>(Wq, wqh, EE * EE);
  f32_to_f16_kernel<<<128,  256, 0, stream>>>(Wk, wkh, KVE * EE);
  f32_to_f16_kernel<<<128,  256, 0, stream>>>(Wv, wvh, KVE * EE);
  f32_to_f16_kernel<<<512,  256, 0, stream>>>(Wo, woh, EE * EE);

  // Q = x @ Wq^T   [4096,1024] : (4096/32)*(1024/64) = 2048 waves
  gemm_nt_kernel<0><<<256, 256, 0, stream>>>(xh, wqh, qh, nullptr, nullptr,
                                             MT, EE, EE);
  // K = x @ Wk^T   [4096,256]  : 512 waves
  gemm_nt_kernel<0><<<64, 256, 0, stream>>>(xh, wkh, kh, nullptr, nullptr,
                                            MT, KVE, EE);
  // V^T = (x @ Wv^T)^T  stored [256,4096]
  gemm_nt_kernel<1><<<64, 256, 0, stream>>>(xh, wvh, vt, nullptr, nullptr,
                                            MT, KVE, EE);

  // attention: B*H*(S/16) = 4096 waves, 4 waves/block
  gqa_attn_kernel<<<1024, 128, 0, stream>>>(qh, kh, vt, ctx);

  // out = ctx @ Wo^T + bo  (fp32)
  gemm_nt_kernel<2><<<256, 256, 0, stream>>>(ctx, woh, nullptr, (float*)d_out,
                                             bo, MT, EE, EE);
}
